// OnlineHadamard_21174188769867
// MI455X (gfx1250) — matrix operand order repaired and checked by reading, not compiled, not run on hardware
//
#include <hip/hip_runtime.h>
#include <math.h>

// ---- problem constants (fixed by reference setup_inputs) ----
#define KDIM 172        // hadK dimension
#define MBLK 64         // power-of-two Hadamard block
#define NCOL 11008      // KDIM*MBLK, columns per token
#define NTOK 8192       // tokens
#define AROWS 176       // padded M (11 * 16)
#define AKP 192         // padded K / row stride of bf16 hadK (6 * 32)
#define GTOK 2          // tokens per workgroup
#define NROWS (GTOK * MBLK)  // 128 Tt rows (= WMMA N per WG)
#define TSTRIDE 200     // Tt row stride in bf16 elems: 400B rows (16B aligned, bank-spread)
#define BLOCK 352       // 11 wave32: 1 WHT task/thread, 1 M-tile row/wave

typedef __attribute__((ext_vector_type(16))) __bf16 v16bf;
typedef __attribute__((ext_vector_type(8)))  float  v8f;

struct alignas(16) U128 { unsigned int w[4]; };
union BF16Frag { v16bf v; U128 q[2]; };

// ---- prepass: hadK fp32 [172x172] -> zero-padded bf16 [176x192] in d_ws ----
__global__ void hadK_to_bf16(const float* __restrict__ hadK, __bf16* __restrict__ A) {
    int idx = blockIdx.x * blockDim.x + threadIdx.x;
    if (idx >= AROWS * AKP) return;
    int r = idx / AKP;
    int c = idx - r * AKP;
    float v = (r < KDIM && c < KDIM) ? hadK[r * KDIM + c] : 0.0f;
    A[idx] = (__bf16)v;
}

// ---- main kernel: fused WHT64 + bf16-split WMMA GEMM over hadK ----
__global__ __launch_bounds__(BLOCK) void had_main(const float* __restrict__ x,
                                                  const __bf16* __restrict__ A,
                                                  float* __restrict__ out) {
    extern __shared__ char smem_raw[];
    __bf16* Thi = (__bf16*)smem_raw;                    // [NROWS][TSTRIDE]
    __bf16* Tlo = Thi + NROWS * TSTRIDE;                // [NROWS][TSTRIDE]

    const int tid = threadIdx.x;
    const int b0  = blockIdx.x * GTOK;
    const float scale = 1.0f / sqrtf((float)NCOL);      // folded into stage 1

    // warm hadK-bf16 region into cache (global_prefetch_b8 path)
    if (tid < 128) __builtin_prefetch((const void*)(A + tid * 264), 0, 3);

    // zero the K-padding columns [KDIM, TSTRIDE) of both planes
    for (int i = tid; i < NROWS * (TSTRIDE - KDIM); i += BLOCK) {
        int r = i / (TSTRIDE - KDIM);
        int c = KDIM + (i - r * (TSTRIDE - KDIM));
        Thi[r * TSTRIDE + c] = (__bf16)0.0f;
        Tlo[r * TSTRIDE + c] = (__bf16)0.0f;
    }

    // -------- stage 1: one (token g, block l) fast WHT64 per thread, scaled,
    // -------- split to bf16 hi/lo, stored transposed: Tt[j_global][l]
    if (tid < GTOK * KDIM) {
        int g = (tid >= KDIM) ? 1 : 0;
        int l = tid - g * KDIM;
        const float4* p = (const float4*)(x + (size_t)(b0 + g) * NCOL + l * MBLK);
        float v[64];
        #pragma unroll
        for (int i = 0; i < 16; ++i) {
            float4 q = p[i];
            v[4*i+0] = q.x; v[4*i+1] = q.y; v[4*i+2] = q.z; v[4*i+3] = q.w;
        }
        // 6-stage butterfly: t[j] = sum_i (-1)^popc(i&j) x[i]
        #pragma unroll
        for (int s = 1; s < 64; s <<= 1) {
            #pragma unroll
            for (int i0 = 0; i0 < 64; i0 += 2 * s) {
                #pragma unroll
                for (int j = 0; j < s; ++j) {
                    float a = v[i0 + j], b = v[i0 + j + s];
                    v[i0 + j]     = a + b;
                    v[i0 + j + s] = a - b;
                }
            }
        }
        int rbase = g * MBLK;
        #pragma unroll 8
        for (int j = 0; j < 64; ++j) {
            float val = v[j] * scale;
            __bf16 h  = (__bf16)val;                 // RNE
            __bf16 lo = (__bf16)(val - (float)h);    // exact residual, then RNE
            Thi[(rbase + j) * TSTRIDE + l] = h;
            Tlo[(rbase + j) * TSTRIDE + l] = lo;
        }
    }
    __syncthreads();

    // -------- stage 2: Y[176 x 128] = A_pad[176 x 192] * T_pad[192 x 128]
    // wave w owns M-tile row w (m0 = 16*w); loops the 8 N-tiles.
    const int wave  = tid >> 5;          // 0..10
    const int lane  = tid & 31;
    const int lhalf = lane >> 4;         // 0 | 1
    const int lmod  = lane & 15;
    const int m0    = wave * 16;

    // preload the 6 A fragments for this M-tile row (loop-invariant, L1-hot)
    // 16-bit A 16x32 layout: halves 0..7 at K=k0+8*lhalf, halves 8..15 at +16
    BF16Frag af[6];
    #pragma unroll
    for (int kk = 0; kk < 6; ++kk) {
        const U128* ap = (const U128*)(A + (m0 + lmod) * AKP + kk * 32 + 8 * lhalf);
        af[kk].q[0] = ap[0];
        af[kk].q[1] = ap[2];             // +16 bf16 elements = +2 x 16B
    }

    for (int nt = 0; nt < 8; ++nt) {
        int n0 = nt * 16;
        v8f c = {0.f, 0.f, 0.f, 0.f, 0.f, 0.f, 0.f, 0.f};
        #pragma unroll
        for (int kk = 0; kk < 6; ++kk) {
            int k0 = kk * 32;
            // B fragments (32x16 bf16 layout): lane j reads Tt row (n0+j),
            // halves 0..7 at K = k0 + 8*lhalf, halves 8..15 at +16 -> two b128 LDS loads
            BF16Frag bh, bl;
            const U128* bph = (const U128*)(Thi + (n0 + lmod) * TSTRIDE + k0 + 8 * lhalf);
            bh.q[0] = bph[0]; bh.q[1] = bph[2];
            const U128* bpl = (const U128*)(Tlo + (n0 + lmod) * TSTRIDE + k0 + 8 * lhalf);
            bl.q[0] = bpl[0]; bl.q[1] = bpl[2];

            c = __builtin_amdgcn_wmma_f32_16x16x32_bf16(false, af[kk].v, false, bh.v,
                                                        (short)0, c, false, false);
            c = __builtin_amdgcn_wmma_f32_16x16x32_bf16(false, af[kk].v, false, bl.v,
                                                        (short)0, c, false, false);
        }

        // store C (f32 C/D layout: N = lane%16, M = r + 8*(lane/16))
        int n = n0 + lmod;
        int g = n >> 6, j = n & 63;
        float* op = out + (size_t)(b0 + g) * NCOL + j + (m0 + 8 * lhalf) * MBLK;
        if (m0 + 16 <= KDIM) {
            // waves 0..9: all 16 M rows valid -> unguarded coalesced stores
            #pragma unroll
            for (int r = 0; r < 8; ++r) op[r * MBLK] = c[r];
        } else {
            // wave 10: rows 160..171 valid, 172..175 are padding
            #pragma unroll
            for (int r = 0; r < 8; ++r)
                if (m0 + 8 * lhalf + r < KDIM) op[r * MBLK] = c[r];
        }
    }
}

extern "C" void kernel_launch(void* const* d_in, const int* in_sizes, int n_in,
                              void* d_out, int out_size, void* d_ws, size_t ws_size,
                              hipStream_t stream) {
    const float* x    = (const float*)d_in[0];   // [8192, 11008] fp32
    const float* hadK = (const float*)d_in[1];   // [172, 172] fp32 (+/-1)
    float* out = (float*)d_out;                  // [8192, 11008] fp32
    __bf16* Abf = (__bf16*)d_ws;                 // 176*192*2 = 67584 bytes

    // prepass: convert + pad hadK to bf16
    hadK_to_bf16<<<(AROWS * AKP + 255) / 256, 256, 0, stream>>>(hadK, Abf);

    // fused WHT + WMMA GEMM: 2 tokens per WG, 11 wave32 per WG
    size_t smem = (size_t)2 * NROWS * TSTRIDE * sizeof(__bf16); // 102400 B
    had_main<<<NTOK / GTOK, BLOCK, smem, stream>>>(x, Abf, out);
}
